// DiagonalSSMLayer_62302795596614
// MI455X (gfx1250) — compile-verified, matrix operand order checked by reading, not run on hardware
//
#include <hip/hip_runtime.h>
#include <hip/hip_bf16.h>

typedef __attribute__((ext_vector_type(16))) __bf16 bf16x16;
typedef __attribute__((ext_vector_type(8)))  __bf16 bf16x8;
typedef __attribute__((ext_vector_type(8)))  float  f32x8;

#define D_MODEL 1024
#define STATE   256
#define SEQ     4096
#define BATCH   4
#define CHUNK   128
#define NCHUNK  (SEQ / CHUNK)    // 32
#define MTOT    (BATCH * SEQ)    // 16384

// workspace byte offsets
#define OFF_BW16  0u            // 256*1024 bf16  = 524288 B
#define OFF_CW16  524288u       // 1024*256 bf16  = 524288 B
#define OFF_LAM   1048576u      // 256 f32 (padded to 4 KB)
#define OFF_BU    1052672u      // 16384*256 f32  = 16777216 B (reused in place as local hs)
#define OFF_CLAST 17829888u     // 4*32*256 f32   = 131072 B
#define OFF_CIN   17960960u     // 4*32*256 f32   = 131072 B
#define OFF_HS16  18092032u     // 16384*256 bf16 = 8388608 B  (total ~25.3 MB)

// ---------------------------------------------------------------------------
// Prep: sigmoid(log_lambda); convert B_w, C_w to bf16 once (reused 1024x).
// ---------------------------------------------------------------------------
__global__ void prep_kernel(const float* __restrict__ Bw, const float* __restrict__ Cw,
                            const float* __restrict__ loglam,
                            __bf16* __restrict__ bw16, __bf16* __restrict__ cw16,
                            float* __restrict__ lam) {
    int tid = blockIdx.x * blockDim.x + threadIdx.x;
    if (tid < D_MODEL * STATE) {
        bw16[tid] = (__bf16)Bw[tid];
        cw16[tid] = (__bf16)Cw[tid];
    }
    if (tid < STATE) lam[tid] = 1.0f / (1.0f + __expf(-loglam[tid]));
}

// ---------------------------------------------------------------------------
// GEMM1: Bu[m,n] = sum_k u[m,k] * B_w[n,k]   (M=16384, N=256, K=1024)
// Each wave computes a 16x64 tile (4 accumulators). A sched_barrier pins all
// 12 loads of a K-step before the first WMMA so they issue as one clause and
// drain with staged s_wait_loadcnt while the 4 WMMAs execute.
// ---------------------------------------------------------------------------
__global__ void gemm1_kernel(const float* __restrict__ u, const __bf16* __restrict__ bw16,
                             float* __restrict__ Bu) {
    const int wave = threadIdx.x >> 5;
    const int lane = threadIdx.x & 31;
    const int half = lane >> 4;      // K-half select per ISA A/B fragment layout
    const int lrow = lane & 15;      // M row (A) / N col (B)
    const int m0 = (blockIdx.x * 8 + wave) * 16;
    const int n0 = blockIdx.y * 64;

    const float* aptr = u + (size_t)(m0 + lrow) * D_MODEL + half * 8;
    const __bf16* bptr[4];
    #pragma unroll
    for (int j = 0; j < 4; ++j)
        bptr[j] = bw16 + (size_t)(n0 + j * 16 + lrow) * D_MODEL + half * 8;

    f32x8 acc[4] = {};
    for (int k0 = 0; k0 < D_MODEL; k0 += 32) {
        // ---- load phase: issue everything up front ----
        float4 a0 = *(const float4*)(aptr + k0);
        float4 a1 = *(const float4*)(aptr + k0 + 4);
        float4 a2 = *(const float4*)(aptr + k0 + 16);
        float4 a3 = *(const float4*)(aptr + k0 + 20);
        bf16x8 b0[4], b1[4];
        #pragma unroll
        for (int j = 0; j < 4; ++j) {
            b0[j] = *(const bf16x8*)(bptr[j] + k0);
            b1[j] = *(const bf16x8*)(bptr[j] + k0 + 16);
        }
        // nothing may cross: forces all loads issued before first WMMA,
        // keeping the 4 B buffers simultaneously live in distinct VGPRs.
        __builtin_amdgcn_sched_barrier(0);
        // ---- compute phase ----
        float at[16] = {a0.x,a0.y,a0.z,a0.w, a1.x,a1.y,a1.z,a1.w,
                        a2.x,a2.y,a2.z,a2.w, a3.x,a3.y,a3.z,a3.w};
        bf16x16 afrag;
        #pragma unroll
        for (int i = 0; i < 16; ++i) afrag[i] = (__bf16)at[i];
        #pragma unroll
        for (int j = 0; j < 4; ++j) {
            bf16x16 bfrag;
            #pragma unroll
            for (int i = 0; i < 8; ++i) { bfrag[i] = b0[j][i]; bfrag[i + 8] = b1[j][i]; }
            acc[j] = __builtin_amdgcn_wmma_f32_16x16x32_bf16(false, afrag, false, bfrag,
                                                             (short)0, acc[j], false, false);
        }
    }
    #pragma unroll
    for (int j = 0; j < 4; ++j) {
        #pragma unroll
        for (int g = 0; g < 8; ++g) {
            const int m = m0 + g + 8 * half;      // C/D layout: VGPR g -> M=g (+8 for hi lanes)
            Bu[(size_t)m * STATE + n0 + j * 16 + lrow] = acc[j][g];
        }
    }
}

// ---------------------------------------------------------------------------
// Scan phase 1: per-(batch,chunk) local recurrence, in place over Bu.
// ---------------------------------------------------------------------------
__global__ void scan_local_kernel(float* __restrict__ Bu, const float* __restrict__ lam,
                                  float* __restrict__ clast) {
    const int b = blockIdx.x / NCHUNK;
    const int c = blockIdx.x % NCHUNK;
    const int n = threadIdx.x;
    const float l = lam[n];
    float h = 0.0f;
    size_t idx = (size_t)(b * SEQ + c * CHUNK) * STATE + n;
    for (int t = 0; t < CHUNK; ++t) {
        h = fmaf(l, h, Bu[idx]);
        Bu[idx] = h;                  // overwrite Bu with carry-free local hs
        idx += STATE;
    }
    clast[(b * NCHUNK + c) * STATE + n] = h;
}

// ---------------------------------------------------------------------------
// Scan phase 2: combine 32 chunk carries per (batch, channel) with lambda^128.
// ---------------------------------------------------------------------------
__global__ void scan_combine_kernel(const float* __restrict__ clast,
                                    const float* __restrict__ lam,
                                    float* __restrict__ cin) {
    const int b = blockIdx.x;
    const int n = threadIdx.x;
    float lC = lam[n];
    #pragma unroll
    for (int i = 0; i < 7; ++i) lC *= lC;   // lambda^128
    float carry = 0.0f;
    for (int c = 0; c < NCHUNK; ++c) {
        cin[(b * NCHUNK + c) * STATE + n] = carry;
        carry = fmaf(lC, carry, clast[(b * NCHUNK + c) * STATE + n]);
    }
}

// ---------------------------------------------------------------------------
// Scan phase 3: hs = local + lambda^{t+1} * carry_in; emit bf16 row-major,
// which is exactly the A-operand feed for GEMM2.
// ---------------------------------------------------------------------------
__global__ void scan_fix_kernel(const float* __restrict__ Bu, const float* __restrict__ cin,
                                const float* __restrict__ lam, __bf16* __restrict__ hs16) {
    const int b = blockIdx.x / NCHUNK;
    const int c = blockIdx.x % NCHUNK;
    const int n = threadIdx.x;
    const float l = lam[n];
    float g = cin[(b * NCHUNK + c) * STATE + n];
    size_t idx = (size_t)(b * SEQ + c * CHUNK) * STATE + n;
    for (int t = 0; t < CHUNK; ++t) {
        g *= l;                                   // lambda^{t+1} * carry
        hs16[idx] = (__bf16)(Bu[idx] + g);
        idx += STATE;
    }
}

// ---------------------------------------------------------------------------
// GEMM2: y[m,d] = sum_n hs[m,n] * C_w[d,n] + D[d]*u[m,d]  (M=16384, N=1024, K=256)
// Each wave computes a 16x64 tile; all 10 loads pinned before the 4 WMMAs.
// ---------------------------------------------------------------------------
__global__ void gemm2_kernel(const __bf16* __restrict__ hs16, const __bf16* __restrict__ cw16,
                             const float* __restrict__ u, const float* __restrict__ Dvec,
                             float* __restrict__ y) {
    const int wave = threadIdx.x >> 5;
    const int lane = threadIdx.x & 31;
    const int half = lane >> 4;
    const int lrow = lane & 15;
    const int m0 = (blockIdx.x * 8 + wave) * 16;
    const int n0 = blockIdx.y * 64;               // d_model tile base

    const __bf16* aptr = hs16 + (size_t)(m0 + lrow) * STATE + half * 8;
    const __bf16* bptr[4];
    #pragma unroll
    for (int j = 0; j < 4; ++j)
        bptr[j] = cw16 + (size_t)(n0 + j * 16 + lrow) * STATE + half * 8;

    f32x8 acc[4] = {};
    #pragma unroll
    for (int k0 = 0; k0 < STATE; k0 += 32) {
        // ---- load phase ----
        bf16x8 a0 = *(const bf16x8*)(aptr + k0);
        bf16x8 a1 = *(const bf16x8*)(aptr + k0 + 16);
        bf16x8 b0[4], b1[4];
        #pragma unroll
        for (int j = 0; j < 4; ++j) {
            b0[j] = *(const bf16x8*)(bptr[j] + k0);
            b1[j] = *(const bf16x8*)(bptr[j] + k0 + 16);
        }
        __builtin_amdgcn_sched_barrier(0);
        // ---- compute phase ----
        bf16x16 afrag;
        #pragma unroll
        for (int i = 0; i < 8; ++i) { afrag[i] = a0[i]; afrag[i + 8] = a1[i]; }
        #pragma unroll
        for (int j = 0; j < 4; ++j) {
            bf16x16 bfrag;
            #pragma unroll
            for (int i = 0; i < 8; ++i) { bfrag[i] = b0[j][i]; bfrag[i + 8] = b1[j][i]; }
            acc[j] = __builtin_amdgcn_wmma_f32_16x16x32_bf16(false, afrag, false, bfrag,
                                                             (short)0, acc[j], false, false);
        }
    }
    #pragma unroll
    for (int j = 0; j < 4; ++j) {
        const int d = n0 + j * 16 + lrow;
        const float Dv = Dvec[d];
        #pragma unroll
        for (int g = 0; g < 8; ++g) {
            const int m = m0 + g + 8 * half;
            y[(size_t)m * D_MODEL + d] = acc[j][g] + Dv * u[(size_t)m * D_MODEL + d];
        }
    }
}

// ---------------------------------------------------------------------------
extern "C" void kernel_launch(void* const* d_in, const int* in_sizes, int n_in,
                              void* d_out, int out_size, void* d_ws, size_t ws_size,
                              hipStream_t stream) {
    const float* u      = (const float*)d_in[0];   // [4,4096,1024]
    const float* loglam = (const float*)d_in[1];   // [256]
    const float* Bw     = (const float*)d_in[2];   // [256,1024]
    const float* Cw     = (const float*)d_in[3];   // [1024,256]
    const float* Dvec   = (const float*)d_in[4];   // [1024]
    float* y = (float*)d_out;

    char* ws = (char*)d_ws;
    __bf16* bw16 = (__bf16*)(ws + OFF_BW16);
    __bf16* cw16 = (__bf16*)(ws + OFF_CW16);
    float*  lam  = (float*)(ws + OFF_LAM);
    float*  Bu   = (float*)(ws + OFF_BU);
    float*  clast= (float*)(ws + OFF_CLAST);
    float*  cin  = (float*)(ws + OFF_CIN);
    __bf16* hs16 = (__bf16*)(ws + OFF_HS16);

    prep_kernel<<<(D_MODEL * STATE + 255) / 256, 256, 0, stream>>>(Bw, Cw, loglam,
                                                                   bw16, cw16, lam);
    gemm1_kernel<<<dim3(MTOT / 128, STATE / 64), 256, 0, stream>>>(u, bw16, Bu);
    scan_local_kernel<<<BATCH * NCHUNK, STATE, 0, stream>>>(Bu, lam, clast);
    scan_combine_kernel<<<BATCH, STATE, 0, stream>>>(clast, lam, cin);
    scan_fix_kernel<<<BATCH * NCHUNK, STATE, 0, stream>>>(Bu, cin, lam, hs16);
    gemm2_kernel<<<dim3(MTOT / 128, D_MODEL / 64), 256, 0, stream>>>(hs16, cw16, u, Dvec, y);
}
